// WindowedDepthXAttn_59193239274012
// MI455X (gfx1250) — matrix operand order repaired and checked by reading, not compiled, not run on hardware
//
#include <hip/hip_runtime.h>
#include <hip/hip_bf16.h>

// ---------------------------------------------------------------------------
// WindowedDepthXAttn for MI455X (gfx1250, wave32, WMMA)
// B=2, C=256, H=W=64 (HW=4096), HEADS=8, dh=32, WIN=7 (Ww=49), TZ=10
// All GEMM operands laid out so every WMMA fragment is two b128 loads.
// conv2 input is zero-halo padded so the implicit-GEMM loop is branch-free.
// ---------------------------------------------------------------------------

#define BB    2
#define CC    256
#define HH_   64
#define WW_   64
#define HWN   4096
#define HEADS 8
#define DH    32
#define WIN   7
#define WW49  49
#define TZ    10
#define CH    128        // C/2 for the depth encoder
#define KC2   (9 * CH)   // 1152, conv2 im2col K
#define PW    66         // padded width/height for conv2 input halo

typedef __attribute__((ext_vector_type(16))) _Float16 v16h;
typedef __attribute__((ext_vector_type(8)))  float    v8f;

union HF16 { v16h v; _Float16 h[16]; uint4 q[2]; };
union VF8  { v8f  v; float    f[8];  };
union H8   { _Float16 h[8]; uint4 q; };

// (dy*66 + dx) * 128 for tap = ky*3+kx, dy=ky-1, dx=kx-1 (halves)
__constant__ int conv_off_tab[9] = {-8576, -8448, -8320, -128, 0, 128, 8320, 8448, 8576};

// ---------------------------------------------------------------------------
// 1) per-batch depth normalization: dn=(d-min)/(max-min+1e-8), invz=1/(dn+1e-6)
// ---------------------------------------------------------------------------
__global__ void depth_norm_kernel(const float* __restrict__ d,
                                  float* __restrict__ dn,
                                  float* __restrict__ invz) {
  const int b = blockIdx.x;
  const int t = threadIdx.x;
  __shared__ float smin[256];
  __shared__ float smax[256];
  const float* db = d + (size_t)b * HWN;
  float mn = 1e30f, mx = -1e30f;
  for (int i = t; i < HWN; i += 256) {
    float v = db[i];
    mn = fminf(mn, v);
    mx = fmaxf(mx, v);
  }
  smin[t] = mn; smax[t] = mx;
  __syncthreads();
  for (int s = 128; s > 0; s >>= 1) {
    if (t < s) {
      smin[t] = fminf(smin[t], smin[t + s]);
      smax[t] = fmaxf(smax[t], smax[t + s]);
    }
    __syncthreads();
  }
  const float lo = smin[0];
  const float inv = 1.0f / (smax[0] - lo + 1e-8f);
  for (int i = t; i < HWN; i += 256) {
    float v = (db[i] - lo) * inv;
    dn[(size_t)b * HWN + i]   = v;
    invz[(size_t)b * HWN + i] = 1.0f / (v + 1e-6f);
  }
}

// ---------------------------------------------------------------------------
// layout converts / fills
// ---------------------------------------------------------------------------
__global__ void zero_f16x8_kernel(_Float16* __restrict__ p, int n8) {
  int i = blockIdx.x * 256 + threadIdx.x;
  if (i < n8) ((uint4*)p)[i] = make_uint4(0, 0, 0, 0);
}

// (B,C,HW) f32 -> (B,HW,C) f16   (channel-last for WMMA B fragments)
__global__ void transpose_cn_f16_kernel(const float* __restrict__ src,
                                        _Float16* __restrict__ dst) {
  const int idx = blockIdx.x * 256 + threadIdx.x;   // B*C*HW, n fastest
  const int n = idx & (HWN - 1);
  const int c = (idx >> 12) & (CC - 1);
  const int b = idx >> 20;
  dst[((size_t)(b * HWN + n)) * CC + c] = (_Float16)src[idx];
}

// de_w2 (256,128,3,3) f32 -> (256, 1152) f16 with k = tap*128 + i
__global__ void w2_linearize_kernel(const float* __restrict__ src,
                                    _Float16* __restrict__ dst) {
  const int idx = blockIdx.x * 256 + threadIdx.x;   // 256*1152
  const int i   = idx & (CH - 1);
  const int tap = (idx >> 7) % 9;
  const int m   = idx / KC2;
  dst[idx] = (_Float16)src[((size_t)m * CH + i) * 9 + tap];
}

// plain f32 -> f16
__global__ void f32_to_f16_kernel(const float* __restrict__ src,
                                  _Float16* __restrict__ dst, int n) {
  int i = blockIdx.x * 256 + threadIdx.x;
  if (i < n) dst[i] = (_Float16)src[i];
}

// ---------------------------------------------------------------------------
// 2) conv3x3, 1 -> 128 channels, zero 'SAME' pad, ReLU.
//    Writes channel-last into the zero-halo padded buffer (B,66,66,128) f16.
// ---------------------------------------------------------------------------
__global__ void conv3x3_c1_relu_kernel(const float* __restrict__ dn,
                                       const float* __restrict__ w1,   // (128,1,3,3)
                                       const float* __restrict__ b1,   // (128,)
                                       _Float16* __restrict__ z1p) {   // (B,66,66,128)
  const int idx = blockIdx.x * 256 + threadIdx.x;   // B*HW*128, o fastest
  const int o = idx & (CH - 1);
  const int n = (idx >> 7) & (HWN - 1);
  const int b = idx >> 19;
  const int y = n >> 6, x = n & 63;
  const float* dnb = dn + (size_t)b * HWN;
  float acc = b1[o];
  #pragma unroll
  for (int ky = 0; ky < 3; ++ky) {
    #pragma unroll
    for (int kx = 0; kx < 3; ++kx) {
      int yy = y + ky - 1, xx = x + kx - 1;
      if (yy >= 0 && yy < HH_ && xx >= 0 && xx < WW_)
        acc += w1[o * 9 + ky * 3 + kx] * dnb[yy * WW_ + xx];
    }
  }
  z1p[((size_t)(b * PW + y + 1) * PW + (x + 1)) * CH + o] = (_Float16)fmaxf(acc, 0.0f);
}

// ---------------------------------------------------------------------------
// 3) conv3x3 128 -> 256 as implicit GEMM on WMMA (branch-free via halo pad).
//    A = w2t (256 x 1152) row-major, B gathered from z1p.
//    8 waves/block; each wave owns a 16x64 tile (4 accumulators, A reused 4x).
//    Output zht (B, HW, 256) channel-last f16.
// ---------------------------------------------------------------------------
__global__ void conv3x3_wmma_kernel(const _Float16* __restrict__ w2t,  // (256,1152)
                                    const _Float16* __restrict__ z1p,  // (B,66,66,128)
                                    const float* __restrict__ b2,      // (256,)
                                    _Float16* __restrict__ zht) {      // (B,HW,256)
  const int b    = blockIdx.z;
  const int lane = threadIdx.x & 31;
  const int wave = threadIdx.x >> 5;
  const int mtile = blockIdx.x * 32 + (wave >> 2) * 16;        // gridDim.x = 8
  const int ntile = blockIdx.y * 256 + (wave & 3) * 64;        // gridDim.y = 16

  const int m     = mtile + (lane & 15);
  const int kbA   = (lane >> 4) * 16;
  const int kbB   = (lane >> 4) * 16;
  const int n0    = ntile + (lane & 15);
  const int y     = n0 >> 6;           // ntile % 64 == 0 -> whole wave tile on one row
  const int xbase = n0 & 63;

  const _Float16* arow = w2t + (size_t)m * KC2;
  // per-lane center pointers (interior of the halo) with kbB folded in
  const _Float16* cptr[4];
  {
    const _Float16* zb = z1p + ((size_t)(b * PW + y + 1) * PW + 1) * CH + kbB;
    #pragma unroll
    for (int s = 0; s < 4; ++s) cptr[s] = zb + (size_t)(xbase + 16 * s) * CH;
  }
  __builtin_prefetch(arow, 0, 3);

  VF8 acc[4] = {{}, {}, {}, {}};
  for (int tap = 0; tap < 9; ++tap) {
    const int koff = conv_off_tab[tap];
    #pragma unroll
    for (int kq = 0; kq < 4; ++kq) {
      const int k0   = tap * 128 + kq * 32;
      const int kloc = kq * 32;
      HF16 a;
      a.q[0] = *(const uint4*)(arow + k0 + kbA);
      a.q[1] = *(const uint4*)(arow + k0 + kbA + 8);
      #pragma unroll
      for (int s = 0; s < 4; ++s) {
        HF16 bf;
        const _Float16* p = cptr[s] + koff + kloc;
        bf.q[0] = *(const uint4*)(p);
        bf.q[1] = *(const uint4*)(p + 8);
        acc[s].v = __builtin_amdgcn_wmma_f32_16x16x32_f16(
            false, a.v, false, bf.v, (short)0, acc[s].v, false, false);
      }
    }
  }

  // epilogue: channel-last store -> one b128 per lane per subtile
  const size_t cb   = (size_t)b * HWN * CC;
  const int nloc    = lane & 15;
  const int rowbase = mtile + (lane >> 4) * 8;    // 8 consecutive output channels
  #pragma unroll
  for (int s = 0; s < 4; ++s) {
    H8 st;
    #pragma unroll
    for (int r = 0; r < 8; ++r) st.h[r] = (_Float16)(acc[s].f[r] + b2[rowbase + r]);
    *(uint4*)(zht + cb + (size_t)(ntile + 16 * s + nloc) * CC + rowbase) = st.q;
  }
}

// ---------------------------------------------------------------------------
// 4) 1x1-conv GEMM: C[b](256x4096, channel-major f32) = A(256x256) * Bt[b]
//    Bt is channel-last (B,HW,256) f16.  bias + optional f32 residual.
//    8 waves/block; each wave 16x64 tile (4 accumulators).
// ---------------------------------------------------------------------------
__global__ void gemm_wmma_bias_kernel(const _Float16* __restrict__ A,    // (256,256)
                                      const _Float16* __restrict__ Bt,   // (B,HW,256)
                                      const float* __restrict__ bias,    // (256,)
                                      const float* __restrict__ residual,// (B,256,HW) | null
                                      float* __restrict__ C) {           // (B,256,HW)
  const int b    = blockIdx.z;
  const int lane = threadIdx.x & 31;
  const int wave = threadIdx.x >> 5;
  const int mtile = blockIdx.x * 32 + (wave >> 2) * 16;        // gridDim.x = 8
  const int ntile = blockIdx.y * 256 + (wave & 3) * 64;        // gridDim.y = 16

  const _Float16* Bb = Bt + (size_t)b * HWN * CC;
  const int m   = mtile + (lane & 15);
  const int kbA = (lane >> 4) * 16;
  const int kbB = (lane >> 4) * 16;
  const int n0  = ntile + (lane & 15);

  const _Float16* arow = A + (size_t)m * CC;
  const _Float16* brow[4];
  #pragma unroll
  for (int s = 0; s < 4; ++s) brow[s] = Bb + (size_t)(n0 + 16 * s) * CC + kbB;
  __builtin_prefetch(arow, 0, 3);
  __builtin_prefetch(brow[0], 0, 3);
  __builtin_prefetch(brow[2], 0, 3);

  VF8 acc[4] = {{}, {}, {}, {}};
  #pragma unroll
  for (int k0 = 0; k0 < CC; k0 += 32) {
    HF16 a;
    a.q[0] = *(const uint4*)(arow + k0 + kbA);
    a.q[1] = *(const uint4*)(arow + k0 + kbA + 8);
    #pragma unroll
    for (int s = 0; s < 4; ++s) {
      HF16 bf;
      bf.q[0] = *(const uint4*)(brow[s] + k0);
      bf.q[1] = *(const uint4*)(brow[s] + k0 + 8);
      acc[s].v = __builtin_amdgcn_wmma_f32_16x16x32_f16(
          false, a.v, false, bf.v, (short)0, acc[s].v, false, false);
    }
  }

  const size_t cb = (size_t)b * CC * HWN;
  const int nloc  = lane & 15;
  const int mhi   = (lane >> 4) * 8;
  #pragma unroll
  for (int r = 0; r < 8; ++r) {
    const int row = mtile + r + mhi;
    const float bs = bias[row];
    #pragma unroll
    for (int s = 0; s < 4; ++s) {
      size_t off = cb + (size_t)row * HWN + ntile + 16 * s + nloc;
      float v = acc[s].f[r] + bs;
      if (residual) v += residual[off];
      C[off] = v;
    }
  }
}

// ---------------------------------------------------------------------------
// 5) windowed attention with depth-bucket bias + softmax.
//    one thread per (b, head, pixel).  Writes channel-last (B,HW,256) f16.
// ---------------------------------------------------------------------------
__device__ __forceinline__ int refl64(int i) {
  i = (i < 0) ? -i : i;                 // reflect (no edge repeat), r=3
  return (i > 63) ? (126 - i) : i;
}

__global__ void win_attn_kernel(const float* __restrict__ q,     // (B,256,HW)
                                const float* __restrict__ kbuf,  // (B,256,HW)
                                const float* __restrict__ vbuf,  // (B,256,HW)
                                const float* __restrict__ invz,  // (B,HW)
                                const float* __restrict__ btab,  // (8,7,7,10)
                                _Float16* __restrict__ out16t) { // (B,HW,256)
  const int idx = blockIdx.x * blockDim.x + threadIdx.x;  // B*HEADS*HW
  const int n  = idx & (HWN - 1);
  const int hh = (idx >> 12) & (HEADS - 1);
  const int b  = idx >> 15;
  const int bh = b * HEADS + hh;

  const float edges[9] = {-0.2f, -0.1f, -0.05f, -0.02f, 0.0f,
                          0.02f, 0.05f, 0.1f, 0.2f};
  const float scale = 0.1767766952966369f;   // 1/sqrt(32)

  const float* qp = q + (size_t)bh * DH * HWN + n;
  float qr[DH];
  #pragma unroll
  for (int dd = 0; dd < DH; ++dd) qr[dd] = qp[(size_t)dd * HWN];

  const int y = n >> 6, x = n & 63;
  const float* ivb = invz + (size_t)b * HWN;
  const float ivc = ivb[n];
  const float* kb = kbuf + (size_t)bh * DH * HWN;
  const float* vb = vbuf + (size_t)bh * DH * HWN;

  float l[WW49];
  float mx = -1e30f;
  for (int w = 0; w < WW49; ++w) {
    const int yy = refl64(y + (w / WIN) - 3);
    const int xx = refl64(x + (w % WIN) - 3);
    const int nn = yy * WW_ + xx;
    float dot = 0.0f;
    #pragma unroll
    for (int dd = 0; dd < DH; ++dd) dot += qr[dd] * kb[(size_t)dd * HWN + nn];
    const float dz = ivb[nn] - ivc;
    int t = 0;
    #pragma unroll
    for (int e = 0; e < 9; ++e) t += (dz >= edges[e]) ? 1 : 0;   // 0..9 == TZ-1
    l[w] = dot * scale + btab[(hh * WW49 + w) * TZ + t];
    mx = fmaxf(mx, l[w]);
  }
  float s = 0.0f;
  for (int w = 0; w < WW49; ++w) { l[w] = __expf(l[w] - mx); s += l[w]; }
  const float rs = 1.0f / s;

  float acc[DH];
  #pragma unroll
  for (int dd = 0; dd < DH; ++dd) acc[dd] = 0.0f;
  for (int w = 0; w < WW49; ++w) {
    const int yy = refl64(y + (w / WIN) - 3);
    const int xx = refl64(x + (w % WIN) - 3);
    const int nn = yy * WW_ + xx;
    const float aw = l[w] * rs;
    #pragma unroll
    for (int dd = 0; dd < DH; ++dd) acc[dd] += aw * vb[(size_t)dd * HWN + nn];
  }
  // channel-last, 32 contiguous halves -> b128 stores
  H8 st;
  _Float16* op = out16t + ((size_t)(b * HWN + n)) * CC + hh * DH;
  #pragma unroll
  for (int g = 0; g < 4; ++g) {
    #pragma unroll
    for (int r = 0; r < 8; ++r) st.h[r] = (_Float16)acc[g * 8 + r];
    *(uint4*)(op + g * 8) = st.q;
  }
}

// ---------------------------------------------------------------------------
// launch
// ---------------------------------------------------------------------------
static inline size_t align256(size_t x) { return (x + 255) & ~(size_t)255; }

extern "C" void kernel_launch(void* const* d_in, const int* in_sizes, int n_in,
                              void* d_out, int out_size, void* d_ws, size_t ws_size,
                              hipStream_t stream) {
  const float* x     = (const float*)d_in[0];    // (B,256,64,64)
  const float* d     = (const float*)d_in[1];    // (B,1,64,64)
  const float* Wq    = (const float*)d_in[2];
  const float* bq    = (const float*)d_in[3];
  const float* Wk    = (const float*)d_in[4];
  const float* bk    = (const float*)d_in[5];
  const float* Wv    = (const float*)d_in[6];
  const float* bv    = (const float*)d_in[7];
  const float* Wo    = (const float*)d_in[8];
  const float* bo    = (const float*)d_in[9];
  const float* de_w1 = (const float*)d_in[10];   // (128,1,3,3)
  const float* de_b1 = (const float*)d_in[11];
  const float* de_w2 = (const float*)d_in[12];   // (256,128,3,3)
  const float* de_b2 = (const float*)d_in[13];
  const float* btab  = (const float*)d_in[14];   // (8,7,7,10)
  float* outp = (float*)d_out;

  // --- carve workspace ---
  char* ws = (char*)d_ws;
  size_t off = 0;
  auto carve = [&](size_t bytes) { void* p = ws + off; off += align256(bytes); return p; };

  float*    dn     = (float*)   carve(sizeof(float)    * BB * HWN);
  float*    invz   = (float*)   carve(sizeof(float)    * BB * HWN);
  _Float16* z1p    = (_Float16*)carve(sizeof(_Float16) * BB * PW * PW * CH);
  _Float16* zht    = (_Float16*)carve(sizeof(_Float16) * BB * HWN * CC);
  _Float16* x16t   = (_Float16*)carve(sizeof(_Float16) * BB * HWN * CC);
  _Float16* out16t = (_Float16*)carve(sizeof(_Float16) * BB * HWN * CC);
  _Float16* w2t    = (_Float16*)carve(sizeof(_Float16) * CC * KC2);
  _Float16* wq16   = (_Float16*)carve(sizeof(_Float16) * CC * CC);
  _Float16* wk16   = (_Float16*)carve(sizeof(_Float16) * CC * CC);
  _Float16* wv16   = (_Float16*)carve(sizeof(_Float16) * CC * CC);
  _Float16* wo16   = (_Float16*)carve(sizeof(_Float16) * CC * CC);
  float*    qf     = (float*)   carve(sizeof(float)    * BB * CC * HWN);
  float*    kf     = (float*)   carve(sizeof(float)    * BB * CC * HWN);
  float*    vf     = (float*)   carve(sizeof(float)    * BB * CC * HWN);
  (void)ws_size; (void)in_sizes; (void)n_in; (void)out_size;

  // --- zero the conv2 halo buffer (border must be 0 every call) ---
  {
    int n8 = BB * PW * PW * CH / 8;
    zero_f16x8_kernel<<<(n8 + 255) / 256, 256, 0, stream>>>(z1p, n8);
  }

  // --- layout conversions (f32 -> f16 WMMA operands) ---
  transpose_cn_f16_kernel<<<BB * CC * HWN / 256, 256, 0, stream>>>(x, x16t);
  w2_linearize_kernel<<<CC * KC2 / 256, 256, 0, stream>>>(de_w2, w2t);
  {
    int n = CC * CC;
    f32_to_f16_kernel<<<(n + 255) / 256, 256, 0, stream>>>(Wq, wq16, n);
    f32_to_f16_kernel<<<(n + 255) / 256, 256, 0, stream>>>(Wk, wk16, n);
    f32_to_f16_kernel<<<(n + 255) / 256, 256, 0, stream>>>(Wv, wv16, n);
    f32_to_f16_kernel<<<(n + 255) / 256, 256, 0, stream>>>(Wo, wo16, n);
  }

  // --- depth normalization ---
  depth_norm_kernel<<<BB, 256, 0, stream>>>(d, dn, invz);

  // --- depth encoder conv1 (1->128, relu), halo-padded channel-last out ---
  conv3x3_c1_relu_kernel<<<BB * HWN * CH / 256, 256, 0, stream>>>(dn, de_w1, de_b1, z1p);

  // --- depth encoder conv2 (128->256) on WMMA, channel-last out ---
  conv3x3_wmma_kernel<<<dim3(CC / 32, HWN / 256, BB), 256, 0, stream>>>(w2t, z1p, de_b2, zht);

  // --- Q = Wq@x + bq ; K = Wk@z + bk ; V = Wv@z + bv  (WMMA GEMMs) ---
  gemm_wmma_bias_kernel<<<dim3(CC / 32, HWN / 256, BB), 256, 0, stream>>>(wq16, x16t, bq, nullptr, qf);
  gemm_wmma_bias_kernel<<<dim3(CC / 32, HWN / 256, BB), 256, 0, stream>>>(wk16, zht,  bk, nullptr, kf);
  gemm_wmma_bias_kernel<<<dim3(CC / 32, HWN / 256, BB), 256, 0, stream>>>(wv16, zht,  bv, nullptr, vf);

  // --- windowed attention + depth-bucket bias + softmax ---
  win_attn_kernel<<<BB * HEADS * HWN / 256, 256, 0, stream>>>(qf, kf, vf, invz, btab, out16t);

  // --- output projection + residual: out = x + Wo@attn + bo ---
  gemm_wmma_bias_kernel<<<dim3(CC / 32, HWN / 256, BB), 256, 0, stream>>>(wo16, out16t, bo, x, outp);
}
